// Attention_39024072852056
// MI455X (gfx1250) — matrix operand order repaired and checked by reading, not compiled, not run on hardware
//
#include <hip/hip_runtime.h>

typedef __bf16 bf16;
typedef __attribute__((ext_vector_type(16))) __bf16 v16bf;
typedef __attribute__((ext_vector_type(8)))  __bf16 v8bf;
typedef __attribute__((ext_vector_type(4)))  __bf16 v4bf;
typedef __attribute__((ext_vector_type(2)))  __bf16 v2bf;
typedef __attribute__((ext_vector_type(8)))  float  v8f;
typedef __attribute__((ext_vector_type(4)))  float  v4f;
typedef __attribute__((ext_vector_type(2)))  float  v2f;

#define NTOK   1024
#define DIM    768
#define HEADS  12
#define HD     64
#define NBH    48          // B * HEADS
#define QK_SCALE 0.125f    // 1/sqrt(64)

__device__ __forceinline__ v8f wmma_bf16(v16bf a, v16bf b, v8f c) {
  // D(16x16,f32) = A(16x32,bf16) * B(32x16,bf16) + C
  return __builtin_amdgcn_wmma_f32_16x16x32_bf16(false, a, false, b, (short)0, c,
                                                 false, false);
}

// A-matrix fragment (16x32 bf16). rowbase = &row[m][0] for this lane's m = lane%16.
// Layout: lanes 0-15 hold K=0..15, lanes 16-31 hold K=16..31 (K = g*16 + 2j + h).
__device__ __forceinline__ v16bf load_a_frag(const bf16* rowbase, int g) {
  v16bf a;
#pragma unroll
  for (int j = 0; j < 8; ++j) {
    int k = (g << 4) + (j << 1);
    v2bf p = *(const v2bf*)(rowbase + k);
    a[2 * j] = p[0];
    a[2 * j + 1] = p[1];
  }
  return a;
}

// B-matrix fragment (32x16 bf16) read from an [N][K]-stored tile.
// rowbase = &tile[n][0] for this lane's n = lane%16.  K = g*16 + 2j + h.
__device__ __forceinline__ v16bf load_b_frag(const bf16* rowbase, int g) {
  v16bf b;
#pragma unroll
  for (int j = 0; j < 8; ++j) {
    int k = (g << 4) + (j << 1);
    v2bf p = *(const v2bf*)(rowbase + k);
    b[2 * j] = p[0];
    b[2 * j + 1] = p[1];
  }
  return b;
}

// B-matrix fragment from a [K][N]-stored tile (leading dim ldk elements).
// colbase = &tile[0][n] for this lane's n = lane%16.  K = g*16 + 2j + h.
__device__ __forceinline__ v16bf load_b_frag_kmajor(const bf16* colbase, int ldk,
                                                    int g) {
  v16bf b;
#pragma unroll
  for (int j = 0; j < 8; ++j) {
    b[2 * j]     = colbase[(size_t)((g << 4) + (j << 1)) * ldk];
    b[2 * j + 1] = colbase[(size_t)((g << 4) + (j << 1) + 1) * ldk];
  }
  return b;
}

// Async global -> LDS copy (16B per lane), tracked with ASYNCcnt.
// LDS offset = low 32 bits of the generic pointer (flat LDS aperture keeps the
// wave-relative LDS byte address in addr[31:0]).
__device__ __forceinline__ void stage_kv_async(const bf16* kSrc, const bf16* vSrc,
                                               bf16* kDst, bf16* vDst) {
  unsigned kOff = (unsigned)(unsigned long long)kDst;
  unsigned vOff = (unsigned)(unsigned long long)vDst;
  asm volatile(
      "global_load_async_to_lds_b128 %0, %2, off\n\t"
      "global_load_async_to_lds_b128 %1, %3, off"
      :: "v"(kOff), "v"(vOff), "v"(kSrc), "v"(vSrc)
      : "memory");
}

// ---------------------------------------------------------------------------
// Kernel 1/4: tiled GEMM  C[M,N] = A[M,768] * W[N,768]^T + bias
// MODE 0: A = f32 x, scatter into Q/K/V bf16 head buffers (K gets QK_SCALE)
// MODE 1: A = bf16 attention output, write f32 d_out
// Block: 256 threads (8 waves as 4x2), tile 128(M) x 64(N), K-step 32.
// ---------------------------------------------------------------------------
template <int MODE>
__global__ void __launch_bounds__(256)
gemm_bf16_kernel(const void* __restrict__ Aptr, const float* __restrict__ Wt,
                 const float* __restrict__ bias,
                 bf16* __restrict__ Qb, bf16* __restrict__ Kb, bf16* __restrict__ Vb,
                 float* __restrict__ Out)
{
  __shared__ __align__(16) bf16 As[128][40];   // [m][k] (+pad)
  __shared__ __align__(16) bf16 Bs[64][40];    // [n][k] (+pad)

  const int tid = threadIdx.x;
  const int lane = tid & 31, wv = tid >> 5;
  const int g = lane >> 4, lm = lane & 15;
  const int wm = wv >> 1, wn = wv & 1;
  const int rowbase = blockIdx.y * 128;
  const int nbase   = blockIdx.x * 64;

  v8f acc[2][2] = {};

  for (int kb = 0; kb < 768; kb += 32) {
    // --- stage A tile (128 x 32) ---
    if (MODE == 0) {
      const float* Af = (const float*)Aptr;
#pragma unroll
      for (int i = 0; i < 4; ++i) {
        int slot = i * 256 + tid;
        int row = slot >> 3, seg = (slot & 7) << 2;
        const float* src = Af + (size_t)(rowbase + row) * 768 + kb + seg;
        if (kb + 32 < 768) __builtin_prefetch(src + 32, 0, 1);
        v4f v = *(const v4f*)src;
        v4bf o;
        o[0] = (bf16)v[0]; o[1] = (bf16)v[1]; o[2] = (bf16)v[2]; o[3] = (bf16)v[3];
        *(v4bf*)(&As[row][seg]) = o;
      }
    } else {
      const bf16* Ab = (const bf16*)Aptr;
#pragma unroll
      for (int i = 0; i < 4; ++i) {
        int slot = i * 256 + tid;
        int row = slot >> 3, seg = (slot & 7) << 2;
        *(v4bf*)(&As[row][seg]) =
            *(const v4bf*)(Ab + (size_t)(rowbase + row) * 768 + kb + seg);
      }
    }
    // --- stage B tile (64 n-rows x 32 k), kept [n][k] ---
#pragma unroll
    for (int i = 0; i < 2; ++i) {
      int slot = i * 256 + tid;
      int n = slot >> 3, seg = (slot & 7) << 2;
      v4f v = *(const v4f*)(Wt + (size_t)(nbase + n) * 768 + kb + seg);
      v4bf o;
      o[0] = (bf16)v[0]; o[1] = (bf16)v[1]; o[2] = (bf16)v[2]; o[3] = (bf16)v[3];
      *(v4bf*)(&Bs[n][seg]) = o;
    }
    __syncthreads();

    v16bf af[2], bfr[2];
#pragma unroll
    for (int am = 0; am < 2; ++am)
      af[am] = load_a_frag(&As[wm * 32 + am * 16 + lm][0], g);
#pragma unroll
    for (int an = 0; an < 2; ++an)
      bfr[an] = load_b_frag(&Bs[wn * 32 + an * 16 + lm][0], g);
#pragma unroll
    for (int am = 0; am < 2; ++am)
#pragma unroll
      for (int an = 0; an < 2; ++an)
        acc[am][an] = wmma_bf16(af[am], bfr[an], acc[am][an]);
    __syncthreads();
  }

  // --- epilogue ---
#pragma unroll
  for (int am = 0; am < 2; ++am)
#pragma unroll
    for (int an = 0; an < 2; ++an)
#pragma unroll
      for (int r = 0; r < 8; ++r) {
        int row = rowbase + wm * 32 + am * 16 + r + (g << 3);
        int col = nbase + wn * 32 + an * 16 + lm;
        float val = acc[am][an][r] + bias[col];
        if (MODE == 0) {
          int s = col / 768, rem = col % 768;
          int head = rem >> 6, d = rem & 63;
          int bidx = row >> 10, ntok = row & 1023;
          size_t dst = ((size_t)(bidx * HEADS + head) << 16) + ((size_t)ntok << 6) + d;
          if (s == 0)      Qb[dst] = (bf16)val;
          else if (s == 1) Kb[dst] = (bf16)(val * QK_SCALE);  // fold softmax scale
          else             Vb[dst] = (bf16)val;
        } else {
          Out[(size_t)row * 768 + col] = val;
        }
      }
}

// ---------------------------------------------------------------------------
// Kernel 2: decomposed relative-position bias.
//   rel_h[bh, qh*32+qw, kh] = sum_c Q[bh, qh*32+qw, c] * rph[qh-kh+31, c]
//   rel_w[bh, qh*32+qw, kw] = sum_c Q[bh, qh*32+qw, c] * rpw[qw-kw+31, c]
// Each block = 1 wave; rel_h fixes qh (rows = qw), rel_w fixes qw (rows = qh)
// so each becomes a dense 32x32x64 GEMM (2x2 WMMA tiles, 2 K-steps).
// ---------------------------------------------------------------------------
__global__ void __launch_bounds__(32)
relpos_kernel(const bf16* __restrict__ Qb, const float* __restrict__ rph,
              const float* __restrict__ rpw,
              float* __restrict__ relh, float* __restrict__ relw)
{
  int task = blockIdx.x;
  const bool isw = task >= NBH * 32;
  if (isw) task -= NBH * 32;
  const int bh = task >> 5, fixed = task & 31;
  const float* rp = isw ? rpw : rph;
  const int lane = threadIdx.x, g = lane >> 4, lm = lane & 15;
  const bf16* Qhead = Qb + ((size_t)bh << 16);

  v8f acc[2][2] = {};
#pragma unroll
  for (int ks = 0; ks < 2; ++ks) {
    v16bf af[2], bfr[2];
#pragma unroll
    for (int am = 0; am < 2; ++am) {
      int m = am * 16 + lm;
      int token = isw ? (m * 32 + fixed) : (fixed * 32 + m);
      af[am] = load_a_frag(Qhead + (size_t)token * 64 + ks * 32, g);
    }
#pragma unroll
    for (int an = 0; an < 2; ++an) {
      int n = an * 16 + lm;                            // output k index
      const float* rrow = rp + (size_t)(fixed - n + 31) * 64 + ks * 32;
      v16bf b;
#pragma unroll
      for (int j = 0; j < 8; ++j) {
        int k = (g << 4) + (j << 1);
        v2f p = *(const v2f*)(rrow + k);
        b[2 * j] = (bf16)p[0];
        b[2 * j + 1] = (bf16)p[1];
      }
      bfr[an] = b;
    }
#pragma unroll
    for (int am = 0; am < 2; ++am)
#pragma unroll
      for (int an = 0; an < 2; ++an)
        acc[am][an] = wmma_bf16(af[am], bfr[an], acc[am][an]);
  }

  float* out = (isw ? relw : relh) + (size_t)bh * (NTOK * 32);
#pragma unroll
  for (int am = 0; am < 2; ++am)
#pragma unroll
    for (int an = 0; an < 2; ++an)
#pragma unroll
      for (int r = 0; r < 8; ++r) {
        int m = am * 16 + r + (g << 3);
        int n = an * 16 + lm;
        int token = isw ? (m * 32 + fixed) : (fixed * 32 + m);
        out[(size_t)token * 32 + n] = acc[am][an][r];
      }
}

// ---------------------------------------------------------------------------
// Kernel 3: fused flash attention with decomposed bias.
// Block: 256 threads = 8 waves; block owns (bh, 128 query rows); each wave
// owns a 16-row q tile. Keys streamed in blocks of 32 (= one kh) with
// DOUBLE-BUFFERED async global->LDS staging (ASYNCcnt), shared by all 8 waves.
// ---------------------------------------------------------------------------
__global__ void __launch_bounds__(256)
attn_kernel(const bf16* __restrict__ Qb, const bf16* __restrict__ Kb,
            const bf16* __restrict__ Vb, const float* __restrict__ relh,
            const float* __restrict__ relw, bf16* __restrict__ AOut)
{
  __shared__ __align__(16) bf16 Ks[2][32][72];    // [buf][key][chan]
  __shared__ __align__(16) bf16 Vs[2][32][72];    // [buf][key][chan]
  __shared__ __align__(16) bf16 Ps[8][16][40];    // per-wave P staging

  const int tid = threadIdx.x, lane = tid & 31, wv = tid >> 5;
  const int g = lane >> 4, lm = lane & 15;
  const int bh = blockIdx.x >> 3;
  const int qbase = ((blockIdx.x & 7) * 8 + wv) * 16;

  const bf16* Qh = Qb + ((size_t)bh << 16);
  const bf16* Kh = Kb + ((size_t)bh << 16);
  const bf16* Vh = Vb + ((size_t)bh << 16);
  const float* rh = relh + (size_t)bh * (NTOK * 32);
  const float* rw = relw + (size_t)bh * (NTOK * 32);

  // Q fragments for this wave's 16 rows (hd=64 -> 2 K-steps), loaded once.
  v16bf qf[2];
#pragma unroll
  for (int ks = 0; ks < 2; ++ks)
    qf[ks] = load_a_frag(Qh + (size_t)(qbase + lm) * 64 + ks * 32, g);

  // rel_w columns are the same (kw=0..31) for every key block: preload.
  float rwv[2][8];
#pragma unroll
  for (int t = 0; t < 2; ++t)
#pragma unroll
    for (int r = 0; r < 8; ++r)
      rwv[t][r] = rw[(size_t)(qbase + r + (g << 3)) * 32 + t * 16 + lm];

  float mrow[8], lrow[8];
  v8f O[4] = {};
#pragma unroll
  for (int r = 0; r < 8; ++r) { mrow[r] = -3.0e38f; lrow[r] = 0.0f; }

  // This thread's staging slice: 16B of K and 16B of V per key block.
  const int skey = tid >> 3, spart = (tid & 7) << 3;

  // Prefetch key block 0 into buffer 0.
  stage_kv_async(Kh + (size_t)(skey) * 64 + spart,
                 Vh + (size_t)(skey) * 64 + spart,
                 &Ks[0][skey][spart], &Vs[0][skey][spart]);

  for (int kb = 0; kb < 32; ++kb) {            // 32 keys per block; kh == kb
    const int cur = kb & 1;
    if (kb < 31) {
      // Prefetch next block into the other buffer, then wait only for the
      // older pair of async transfers (current buffer) to land.
      stage_kv_async(Kh + (size_t)((kb + 1) * 32 + skey) * 64 + spart,
                     Vh + (size_t)((kb + 1) * 32 + skey) * 64 + spart,
                     &Ks[cur ^ 1][skey][spart], &Vs[cur ^ 1][skey][spart]);
      asm volatile("s_wait_asynccnt 2" ::: "memory");
    } else {
      asm volatile("s_wait_asynccnt 0" ::: "memory");
    }
    __syncthreads();                           // current buffer visible to all

    // --- S = Q * K^T (16 x 32 keys, as two 16x16 tiles) ---
    v8f S[2] = {};
#pragma unroll
    for (int t = 0; t < 2; ++t)
#pragma unroll
      for (int ks = 0; ks < 2; ++ks) {
        v16bf bk = load_b_frag(&Ks[cur][t * 16 + lm][ks * 32], g);
        S[t] = wmma_bf16(qf[ks], bk, S[t]);
      }

    // --- add decomposed bias:  rel_h[q, kb] + rel_w[q, kw] ---
    float rhv[8];
#pragma unroll
    for (int r = 0; r < 8; ++r)
      rhv[r] = rh[(size_t)(qbase + r + (g << 3)) * 32 + kb];
#pragma unroll
    for (int t = 0; t < 2; ++t)
#pragma unroll
      for (int r = 0; r < 8; ++r) S[t][r] += rhv[r] + rwv[t][r];

    // --- online softmax (rows live in 16-lane groups) ---
    float mnew[8], alpha[8];
#pragma unroll
    for (int r = 0; r < 8; ++r) {
      float v = fmaxf(S[0][r], S[1][r]);
#pragma unroll
      for (int off = 8; off >= 1; off >>= 1) v = fmaxf(v, __shfl_xor(v, off, 32));
      mnew[r] = fmaxf(mrow[r], v);
      alpha[r] = __expf(mrow[r] - mnew[r]);
      mrow[r] = mnew[r];
    }
#pragma unroll
    for (int r = 0; r < 8; ++r) {
      S[0][r] = __expf(S[0][r] - mnew[r]);
      S[1][r] = __expf(S[1][r] - mnew[r]);
      float s = S[0][r] + S[1][r];
#pragma unroll
      for (int off = 8; off >= 1; off >>= 1) s += __shfl_xor(s, off, 32);
      lrow[r] = lrow[r] * alpha[r] + s;
    }
#pragma unroll
    for (int nb = 0; nb < 4; ++nb)
#pragma unroll
      for (int r = 0; r < 8; ++r) O[nb][r] *= alpha[r];

    // --- P: C-layout -> A-fragment layout via per-wave LDS ---
#pragma unroll
    for (int t = 0; t < 2; ++t)
#pragma unroll
      for (int r = 0; r < 8; ++r)
        Ps[wv][r + (g << 3)][t * 16 + lm] = (bf16)S[t][r];
    v16bf pf = load_a_frag(&Ps[wv][lm][0], g);   // same-wave DScnt dependency

    // --- O += P(16x32) * V(32x64); V is [key][chan] so read K-major ---
#pragma unroll
    for (int nb = 0; nb < 4; ++nb) {
      v16bf bv = load_b_frag_kmajor(&Vs[cur][0][nb * 16 + lm], 72, g);
      O[nb] = wmma_bf16(pf, bv, O[nb]);
    }
    __syncthreads();                           // all waves done with buffer cur
  }

  // --- normalize + write [b, n, head*64+d] bf16 for proj GEMM ---
  const int b = bh / HEADS, head = bh % HEADS;
#pragma unroll
  for (int r = 0; r < 8; ++r) {
    float inv = 1.0f / lrow[r];
    size_t token = (size_t)b * NTOK + qbase + r + (g << 3);
#pragma unroll
    for (int nb = 0; nb < 4; ++nb)
      AOut[token * 768 + head * 64 + nb * 16 + lm] = (bf16)(O[nb][r] * inv);
  }
}

// ---------------------------------------------------------------------------
extern "C" void kernel_launch(void* const* d_in, const int* in_sizes, int n_in,
                              void* d_out, int out_size, void* d_ws, size_t ws_size,
                              hipStream_t stream) {
  (void)in_sizes; (void)n_in; (void)out_size; (void)ws_size;
  const float* x      = (const float*)d_in[0];   // [4,32,32,768]
  const float* qkv_w  = (const float*)d_in[1];   // [2304,768]
  const float* qkv_b  = (const float*)d_in[2];   // [2304]
  const float* proj_w = (const float*)d_in[3];   // [768,768]
  const float* proj_b = (const float*)d_in[4];   // [768]
  const float* rph    = (const float*)d_in[5];   // [63,64]
  const float* rpw    = (const float*)d_in[6];   // [63,64]
  float* out = (float*)d_out;                    // [4,32,32,768]

  char* ws = (char*)d_ws;
  const size_t headBytes = (size_t)NBH * NTOK * HD * sizeof(bf16);   // 6.29 MB
  const size_t relBytes  = (size_t)NBH * NTOK * 32 * sizeof(float);  // 6.29 MB
  bf16*  Qb   = (bf16*)ws;  ws += headBytes;
  bf16*  Kb   = (bf16*)ws;  ws += headBytes;
  bf16*  Vb   = (bf16*)ws;  ws += headBytes;
  float* relh = (float*)ws; ws += relBytes;
  float* relw = (float*)ws; ws += relBytes;
  bf16*  AOut = (bf16*)ws;  ws += (size_t)4 * NTOK * DIM * sizeof(bf16);

  // 1) QKV GEMM + bias, scatter into per-head Q/K/V (scale folded into K)
  gemm_bf16_kernel<0><<<dim3(2304 / 64, 4096 / 128), 256, 0, stream>>>(
      x, qkv_w, qkv_b, Qb, Kb, Vb, nullptr);

  // 2) decomposed relative position bias (rel_h, rel_w)
  relpos_kernel<<<dim3(NBH * 32 * 2), 32, 0, stream>>>(Qb, rph, rpw, relh, relw);

  // 3) fused flash attention with bias (async double-buffered K/V)
  attn_kernel<<<dim3(NBH * 8), 256, 0, stream>>>(Qb, Kb, Vb, relh, relw, AOut);

  // 4) output projection + bias -> f32 d_out
  gemm_bf16_kernel<1><<<dim3(768 / 64, 4096 / 128), 256, 0, stream>>>(
      AOut, proj_w, proj_b, nullptr, nullptr, nullptr, out);
}
